// Encoder_90563680404109
// MI455X (gfx1250) — matrix operand order repaired and checked by reading, not compiled, not run on hardware
//
#include <hip/hip_runtime.h>
#include <stdint.h>

// Problem constants (match reference)
#define TT 256
#define BB 64
#define EE 512
#define HH 512
#define GG (3*HH)   // 1536 gates

typedef __attribute__((ext_vector_type(16))) __bf16   v16bf;
typedef __attribute__((ext_vector_type(8)))  float    v8f;
typedef __attribute__((ext_vector_type(4)))  uint32_t v4u;
typedef unsigned short u16;

union ABu { v4u u[2]; v16bf v; };

// ---- helpers ----------------------------------------------------------------

__device__ __forceinline__ u16 f2bf(float f) {
  uint32_t x = __float_as_uint(f);
  // round-to-nearest-even f32 -> bf16
  uint32_t r = (x + 0x7fffu + ((x >> 16) & 1u)) >> 16;
  return (u16)r;
}

__device__ __forceinline__ float sigmoidf(float x) {
  return 1.0f / (1.0f + __expf(-x));
}

__device__ __forceinline__ v8f wmma_bf16(const ABu& a, const ABu& b, v8f c) {
  return __builtin_amdgcn_wmma_f32_16x16x32_bf16(
      /*neg_a=*/false, a.v, /*neg_b=*/false, b.v,
      /*c_mod=*/(short)0, c, /*reuse_a=*/false, /*reuse_b=*/false);
}

// One wave computes a 32x32 f32 tile (2x2 register blocking) of
// C = A(bf16,row-major) x B(bf16,row-major) + bias.
// A: M x K, lda = K. B: K x N, ldb = N. C: M x N f32, ldc = N. K mult of 32.
// All fragment pointers advance additively (no per-iteration 64-bit multiply).
__device__ __forceinline__ void wmma_tile32(const u16* __restrict__ A, int lda,
                                            const u16* __restrict__ B, int ldb,
                                            const float* __restrict__ bias,
                                            float* __restrict__ C, int ldc,
                                            int mbase, int nbase, int K) {
  const int lane = threadIdx.x;      // 0..31
  const int hl   = lane >> 4;        // 0 or 1
  const int l16  = lane & 15;

  v8f acc00 = {}, acc01 = {}, acc10 = {}, acc11 = {};

  // A fragment pointers: row (mbase+l16) and (mbase+16+l16), K-chunk by lane half
  const u16* pa0 = A + (size_t)(mbase + l16) * lda + hl * 8;
  const u16* pa1 = pa0 + (size_t)16 * lda;
  // B fragment pointer: lane holds column nbase+l16, K-chunk by lane half
  const u16* pb  = B + (size_t)(hl * 8) * ldb + nbase + l16;
  const size_t bstep = (size_t)32 * ldb;

  #pragma unroll 2
  for (int k = 0; k < K; k += 32) {
    ABu a0, a1, b0, b1;
    // A 16x32 fragments, ISA lane/K layout (16-byte aligned chunks)
    a0.u[0] = *(const v4u*)(pa0);
    a0.u[1] = *(const v4u*)(pa0 + 16);
    a1.u[0] = *(const v4u*)(pa1);
    a1.u[1] = *(const v4u*)(pa1 + 16);
    // B 32x16 fragments: strided column gather, ISA lane/K layout (two N-tiles)
    #pragma unroll
    for (int s = 0; s < 16; s += 2) {
      const int kk = (s < 8) ? s : (8 + s);
      const u16* q = pb + (size_t)kk * ldb;
      b0.u[s >> 3][(s >> 1) & 3] = (uint32_t)q[0] | ((uint32_t)q[ldb] << 16);
      b1.u[s >> 3][(s >> 1) & 3] = (uint32_t)q[16] | ((uint32_t)q[ldb + 16] << 16);
    }
    // 4 WMMA per 8 loads (2x fragment reuse each way)
    acc00 = wmma_bf16(a0, b0, acc00);
    acc01 = wmma_bf16(a0, b1, acc01);
    acc10 = wmma_bf16(a1, b0, acc10);
    acc11 = wmma_bf16(a1, b1, acc11);
    pa0 += 32;
    pa1 += 32;
    pb  += bstep;
  }

  const float bv0 = bias[nbase + l16];
  const float bv1 = bias[nbase + 16 + l16];
  #pragma unroll
  for (int v = 0; v < 8; ++v) {
    const size_t r0 = (size_t)(mbase + v + hl * 8) * ldc;
    const size_t r1 = (size_t)(mbase + 16 + v + hl * 8) * ldc;
    C[r0 + nbase + l16]      = acc00[v] + bv0;
    C[r0 + nbase + 16 + l16] = acc01[v] + bv1;
    C[r1 + nbase + l16]      = acc10[v] + bv0;
    C[r1 + nbase + 16 + l16] = acc11[v] + bv1;
  }
}

// ---- kernels ----------------------------------------------------------------

// Dual-direction GEMM: blockIdx.y selects {A,B,bias,C} set. One wave = 32x32 tile.
__global__ void gemm_wmma_kernel(const u16* A0, const u16* A1,
                                 const u16* B0, const u16* B1,
                                 const float* bias0, const float* bias1,
                                 float* C0, float* C1,
                                 int M, int N, int K) {
  const u16*   A    = blockIdx.y ? A1    : A0;
  const u16*   B    = blockIdx.y ? B1    : B0;
  const float* bias = blockIdx.y ? bias1 : bias0;
  float*       C    = blockIdx.y ? C1    : C0;

  const int ntN  = N >> 5;                        // 32-wide tiles in N
  const int tile = blockIdx.x * blockDim.y + threadIdx.y;
  const int mt   = tile / ntN;
  const int nt   = tile % ntN;
  if ((mt << 5) >= M) return;                     // wave-uniform guard
  wmma_tile32(A, K, B, N, bias, C, N, mt << 5, nt << 5, K);
}

// Embedding gather (+ per-batch sequence reversal for backward dir) -> bf16
__global__ void embed_kernel(const int* __restrict__ src,
                             const int* __restrict__ len_src,
                             const float* __restrict__ embed_w,
                             u16* __restrict__ embf, u16* __restrict__ embb) {
  const int tb = blockIdx.x;            // 0..T*B-1
  const int t = tb / BB, b = tb % BB;
  const int tokf = src[t * BB + b];
  const int L = len_src[b];
  const int tr = (t < L) ? (L - 1 - t) : t;
  const int tokb = src[tr * BB + b];
  const float* rf = embed_w + (size_t)tokf * EE;
  const float* rb = embed_w + (size_t)tokb * EE;
  u16* of = embf + (size_t)tb * EE;
  u16* ob = embb + (size_t)tb * EE;
  for (int e = threadIdx.x; e < EE; e += blockDim.x) {
    of[e] = f2bf(rf[e]);
    ob[e] = f2bf(rb[e]);
  }
}

// W [rows][cols] f32 -> WT [cols][rows] bf16  (puts weights in B row-major form)
__global__ void transpose_bf16_kernel(const float* __restrict__ W,
                                      u16* __restrict__ WT, int rows, int cols) {
  const int i = blockIdx.x * blockDim.x + threadIdx.x;
  if (i >= rows * cols) return;
  const int g = i / cols, e = i % cols;
  WT[(size_t)e * rows + g] = f2bf(W[i]);
}

__global__ void init_h_kernel(float* __restrict__ h, u16* __restrict__ hbf) {
  const int i = blockIdx.x * blockDim.x + threadIdx.x;
  if (i >= 2 * BB * HH) return;
  h[i] = 0.0f;
  hbf[i] = 0;
}

// GRU gate fusion for one timestep, both directions (idx dim = 2*B*H)
__global__ void gru_update_kernel(const float* __restrict__ xg0,
                                  const float* __restrict__ xg1,
                                  const float* __restrict__ hg,
                                  float* __restrict__ h,
                                  u16* __restrict__ hbf,
                                  float* __restrict__ out, int t) {
  const int idx = blockIdx.x * blockDim.x + threadIdx.x;
  if (idx >= 2 * BB * HH) return;
  const int d = idx / (BB * HH);
  const int r = idx % (BB * HH);
  const int b = r / HH, j = r % HH;

  const float* xg = d ? xg1 : xg0;
  const size_t xrow = ((size_t)t * BB + b) * GG;
  const float xr = xg[xrow + j];
  const float xz = xg[xrow + HH + j];
  const float xn = xg[xrow + 2 * HH + j];

  const size_t hrow = ((size_t)d * BB + b) * GG;
  const float hr = hg[hrow + j];
  const float hz = hg[hrow + HH + j];
  const float hn = hg[hrow + 2 * HH + j];

  const float rg = sigmoidf(xr + hr);
  const float z  = sigmoidf(xz + hz);
  const float n  = tanhf(xn + rg * hn);

  const size_t hidx = ((size_t)d * BB + b) * HH + j;
  const float hnew = (1.0f - z) * n + z * h[hidx];
  h[hidx]  = hnew;
  hbf[hidx] = f2bf(hnew);
  // outputs[t, b, d*H + j]
  out[(size_t)t * BB * 2 * HH + (size_t)b * 2 * HH + (size_t)d * HH + j] = hnew;
}

// hidden[b, j] = out_b[len_src[b]-1, b, j]  (read back from d_out)
__global__ void hidden_kernel(const int* __restrict__ len_src,
                              const float* __restrict__ outbuf,
                              float* __restrict__ hid) {
  const int i = blockIdx.x * blockDim.x + threadIdx.x;
  if (i >= BB * HH) return;
  const int b = i / HH, j = i % HH;
  const int t = len_src[b] - 1;
  hid[i] = outbuf[(size_t)t * BB * 2 * HH + (size_t)b * 2 * HH + HH + j];
}

// ---- launch -----------------------------------------------------------------

extern "C" void kernel_launch(void* const* d_in, const int* in_sizes, int n_in,
                              void* d_out, int out_size, void* d_ws, size_t ws_size,
                              hipStream_t stream) {
  const int*   src     = (const int*)  d_in[0];
  const int*   len_src = (const int*)  d_in[1];
  const float* embed_w = (const float*)d_in[2];
  const float* Wih_f   = (const float*)d_in[3];
  const float* Whh_f   = (const float*)d_in[4];
  const float* bih_f   = (const float*)d_in[5];
  const float* bhh_f   = (const float*)d_in[6];
  const float* Wih_b   = (const float*)d_in[7];
  const float* Whh_b   = (const float*)d_in[8];
  const float* bih_b   = (const float*)d_in[9];
  const float* bhh_b   = (const float*)d_in[10];
  float* out = (float*)d_out;

  // workspace carve-up (all offsets 256B-aligned)
  char* ws = (char*)d_ws;
  size_t o = 0;
  u16* embf   = (u16*)(ws + o); o += (size_t)TT * BB * EE * 2;   // 16 MB
  u16* embb   = (u16*)(ws + o); o += (size_t)TT * BB * EE * 2;   // 16 MB
  u16* wihT_f = (u16*)(ws + o); o += (size_t)EE * GG * 2;        // 1.5 MB
  u16* wihT_b = (u16*)(ws + o); o += (size_t)EE * GG * 2;
  u16* whhT_f = (u16*)(ws + o); o += (size_t)HH * GG * 2;
  u16* whhT_b = (u16*)(ws + o); o += (size_t)HH * GG * 2;
  float* xg_f = (float*)(ws + o); o += (size_t)TT * BB * GG * 4; // 100 MB
  float* xg_b = (float*)(ws + o); o += (size_t)TT * BB * GG * 4; // 100 MB
  float* hg   = (float*)(ws + o); o += (size_t)2 * BB * GG * 4;  // hg[2][B][3H]
  float* h    = (float*)(ws + o); o += (size_t)2 * BB * HH * 4;  // h[2][B][H]
  u16*  hbf   = (u16*) (ws + o); o += (size_t)2 * BB * HH * 2;

  // 0) h = 0
  init_h_kernel<<<dim3((2 * BB * HH + 255) / 256), dim3(256), 0, stream>>>(h, hbf);

  // 1) weights -> transposed bf16  (B-operand row-major [K][N])
  {
    const int n = GG * EE;
    dim3 g((n + 255) / 256), blk(256);
    transpose_bf16_kernel<<<g, blk, 0, stream>>>(Wih_f, wihT_f, GG, EE);
    transpose_bf16_kernel<<<g, blk, 0, stream>>>(Wih_b, wihT_b, GG, EE);
    transpose_bf16_kernel<<<g, blk, 0, stream>>>(Whh_f, whhT_f, GG, HH);
    transpose_bf16_kernel<<<g, blk, 0, stream>>>(Whh_b, whhT_b, GG, HH);
  }

  // 2) embedding gather (+reversal for backward) -> bf16
  embed_kernel<<<dim3(TT * BB), dim3(128), 0, stream>>>(src, len_src, embed_w, embf, embb);

  // 3) big input GEMM (both dirs): xg = emb @ WihT + bih   [16384 x 1536], K=512
  {
    const int M = TT * BB, N = GG, K = EE;
    const int tiles = (M / 32) * (N / 32);        // 24576, divisible by 8
    gemm_wmma_kernel<<<dim3(tiles / 8, 2), dim3(32, 8), 0, stream>>>(
        embf, embb, wihT_f, wihT_b, bih_f, bih_b, xg_f, xg_b, M, N, K);
  }

  // 4) recurrence: 256 steps, each = small WMMA GEMM + fused gate update
  for (int t = 0; t < TT; ++t) {
    const int M = BB, N = GG, K = HH;
    const int tiles = (M / 32) * (N / 32);        // 96, divisible by 8 -> 12 blocks
    gemm_wmma_kernel<<<dim3(tiles / 8, 2), dim3(32, 8), 0, stream>>>(
        hbf, hbf + (size_t)BB * HH,
        whhT_f, whhT_b, bhh_f, bhh_b,
        hg, hg + (size_t)BB * GG, M, N, K);
    gru_update_kernel<<<dim3((2 * BB * HH + 255) / 256), dim3(256), 0, stream>>>(
        xg_f, xg_b, hg, h, hbf, out, t);
  }

  // 5) hidden = out_b[len-1]
  hidden_kernel<<<dim3((BB * HH + 255) / 256), dim3(256), 0, stream>>>(
      len_src, out, out + (size_t)TT * BB * 2 * HH);
}
